// CompQGCNEncoder_88252987998402
// MI455X (gfx1250) — matrix operand order repaired and checked by reading, not compiled, not run on hardware
//
#include <hip/hip_runtime.h>
#include <hip/hip_bf16.h>

// ---------------- Problem constants ----------------
#define NUM_ENT 40000
#define NREL2   200          // 2*NUM_REL
#define DD      200          // feature dim
#define EDG     80000        // edges per side
#define TOTE    160000       // 2*EDG
#define BQ      1024
#define KPAD    224          // 7 * 32
#define NPAD    208          // 13 * 16
#define NTILES  13
#define KTILES  7
#define CHUNKS  56           // KPAD / 4 float4-chunks per row (data chunks: 50)
#define LDA     232          // LDS row stride in bf16 elems (29*8: 16B-aligned, odd multiple of 16B)
#define MT      4            // M-tiles per block (64 rows)
#define ROWS    64
#define BN_EPS  1e-5f

typedef __attribute__((ext_vector_type(16))) __bf16 v16bf;
typedef __attribute__((ext_vector_type(8)))  float  v8f;

union V16U { v16bf v; uint4 q[2]; };

// A fragment (16x32 bf16): lane L holds row M=L&15; lanes 0-15 cover K={k0..k0+7, k0+16..k0+23},
// lanes 16-31 cover K={k0+8..k0+15, k0+24..k0+31}  -> two contiguous 16-byte LDS loads per lane.
__device__ __forceinline__ v16bf load_a_frag(const __bf16* base, int lane, int k0) {
    int l16 = lane & 15, hf = lane >> 4;
    const __bf16* p = base + l16 * LDA + k0 + hf * 8;
    V16U t;
    t.q[0] = *reinterpret_cast<const uint4*>(p);
    t.q[1] = *reinterpret_cast<const uint4*>(p + 16);
    return t.v;
}

// B fragment (32x16 bf16) from transposed weight Wt[N=208][K=224]:
// lane L holds column N = nt*16 + (L&15); K = {k0+8*(L>>4)..+7, k0+16+8*(L>>4)..+7} (two 16-byte loads).
__device__ __forceinline__ v16bf load_b_frag(const __bf16* wt, int nt, int lane, int k0) {
    int l16 = lane & 15, hf = lane >> 4;
    const __bf16* p = wt + (size_t)(nt * 16 + l16) * KPAD + k0 + hf * 8;
    V16U t;
    t.q[0] = *reinterpret_cast<const uint4*>(p);
    t.q[1] = *reinterpret_cast<const uint4*>(p + 16);
    return t.v;
}

__device__ __forceinline__ v8f wmma_bf16(v16bf a, v16bf b, v8f c) {
    return __builtin_amdgcn_wmma_f32_16x16x32_bf16(false, a, false, b, (short)0, c, false, false);
}

__device__ __forceinline__ float4 load_f4(const float* p) {
    return *reinterpret_cast<const float4*>(p);
}
__device__ __forceinline__ void store_bf4(__bf16* dst, float a, float b, float c, float d) {
    union { __bf16 h[4]; uint2 u; } t;
    t.h[0] = (__bf16)a; t.h[1] = (__bf16)b; t.h[2] = (__bf16)c; t.h[3] = (__bf16)d;
    *reinterpret_cast<uint2*>(dst) = t.u;
}
__device__ __forceinline__ float4 load_bf4f(const __bf16* p) {
    union { __bf16 h[4]; uint2 u; } t;
    t.u = *reinterpret_cast<const uint2*>(p);
    return make_float4((float)t.h[0], (float)t.h[1], (float)t.h[2], (float)t.h[3]);
}

// ---------------- degree / dinv ----------------
__global__ void deg_kernel(const int* __restrict__ ei0, float* __restrict__ deg_in,
                           float* __restrict__ deg_out, int e) {
    int i = blockIdx.x * blockDim.x + threadIdx.x;
    if (i >= 2 * e) return;
    int s = ei0[i];
    atomicAdd((i < e ? deg_in : deg_out) + s, 1.0f);
}

__global__ void dinv_kernel(const float* __restrict__ deg_in, const float* __restrict__ deg_out,
                            float* __restrict__ dinv_in, float* __restrict__ dinv_out, int n) {
    int i = blockIdx.x * blockDim.x + threadIdx.x;
    if (i >= n) return;
    float a = deg_in[i], b = deg_out[i];
    dinv_in[i]  = (a > 0.f) ? rsqrtf(a) : 0.f;
    dinv_out[i] = (b > 0.f) ? rsqrtf(b) : 0.f;
}

// ---------------- weight transpose + bf16 convert: Wt[n][k] = W[k][n] ----------------
__global__ void wprep_kernel(const float* __restrict__ w, __bf16* __restrict__ wt) {
    int i = blockIdx.x * blockDim.x + threadIdx.x;
    if (i >= NPAD * KPAD) return;
    int n = i / KPAD, k = i % KPAD;
    float v = (n < DD && k < DD) ? w[(size_t)k * DD + n] : 0.f;
    wt[i] = (__bf16)v;
}

// ---------------- fused edge-message kernel: 64 edges / block ----------------
__global__ __launch_bounds__(128) void edge_msg_kernel(
    const float* __restrict__ x, const float* __restrict__ relc,
    const int* __restrict__ src, const int* __restrict__ dst, const int* __restrict__ et,
    const int* __restrict__ qe, const int* __restrict__ qr, int tot,
    const float* __restrict__ dinv, const __bf16* __restrict__ wqT, const __bf16* __restrict__ wT,
    float* __restrict__ acc)
{
    __shared__ __align__(16) __bf16 bufA[ROWS * LDA];   // qsum (GEMM1 A)
    __shared__ __align__(16) __bf16 bufB[ROWS * LDA];   // a2   (GEMM2 A)
    __shared__ __align__(16) __bf16 bufQ[ROWS * LDA];   // qual (GEMM1 D, bf16)
    __shared__ int   s_src[ROWS];
    __shared__ int   s_dst[ROWS];
    __shared__ int   s_et [ROWS];
    __shared__ float s_nrm[ROWS];

    const int tid  = threadIdx.x;
    const int wid  = tid >> 5;
    const int lane = tid & 31;
    const int ebase = blockIdx.x * ROWS;           // grid exact: EDG/64 blocks

    if (tid < ROWS) {
        int ge = ebase + tid;
        int s = src[ge], d2 = dst[ge];
        s_src[tid] = s;
        s_dst[tid] = d2;
        s_et [tid] = et[ge];
        s_nrm[tid] = dinv[s] * dinv[d2];
    }

    // Stage 1 (float4): qsum[r][d] = sum_q x[qe]*rel[qr] -> bf16 A operand (zero K-pad folded in)
    for (int rr = 0; rr < 16; ++rr) {
        int r  = wid * 16 + rr;
        int ge = ebase + r;
        const float* xa = x    + (size_t)qe[ge]       * DD;
        const float* xb = x    + (size_t)qe[tot + ge] * DD;
        const float* ra = relc + (size_t)qr[ge]       * DD;
        const float* rb = relc + (size_t)qr[tot + ge] * DD;
        for (int c = lane; c < CHUNKS; c += 32) {
            int d = c * 4;
            float vx = 0.f, vy = 0.f, vz = 0.f, vw = 0.f;
            if (d < DD) {
                float4 a4 = load_f4(xa + d), b4 = load_f4(xb + d);
                float4 p4 = load_f4(ra + d), q4 = load_f4(rb + d);
                vx = a4.x * p4.x + b4.x * q4.x;
                vy = a4.y * p4.y + b4.y * q4.y;
                vz = a4.z * p4.z + b4.z * q4.z;
                vw = a4.w * p4.w + b4.w * q4.w;
            }
            store_bf4(bufA + r * LDA + d, vx, vy, vz, vw);
        }
    }
    __syncthreads();

    const int l16 = lane & 15, hf = lane >> 4;

    // GEMM1: qual = qsum @ w_q ; waves own N-tiles {wid, wid+4, ...}; B double-buffered,
    // reused across 4 M-tiles; D written to bufQ as bf16 (no gathers in this loop).
    for (int nt = wid; nt < NTILES; nt += 4) {
        v8f c[MT];
        #pragma unroll
        for (int m = 0; m < MT; ++m) c[m] = v8f{0.f, 0.f, 0.f, 0.f, 0.f, 0.f, 0.f, 0.f};
        v16bf bcur = load_b_frag(wqT, nt, lane, 0);
        #pragma unroll
        for (int kt = 0; kt < KTILES; ++kt) {
            v16bf bnext;
            if (kt + 1 < KTILES) bnext = load_b_frag(wqT, nt, lane, (kt + 1) * 32);
            #pragma unroll
            for (int m = 0; m < MT; ++m)
                c[m] = wmma_bf16(load_a_frag(bufA + m * 16 * LDA, lane, kt * 32), bcur, c[m]);
            if (kt + 1 < KTILES) bcur = bnext;
        }
        int col = nt * 16 + l16;          // col <= 207 < LDA: unconditional store is safe
        #pragma unroll
        for (int m = 0; m < MT; ++m) {
            #pragma unroll
            for (int v = 0; v < 8; ++v) {
                int rr = m * 16 + v + 8 * hf;
                bufQ[rr * LDA + col] = (__bf16)c[m][v];
            }
        }
    }
    __syncthreads();

    // Stage 3 (float4): a2 = x[src] * (0.5*rel[etype] + 0.5*qual) -> bufB (zero K-pad folded in)
    for (int rr = 0; rr < 16; ++rr) {
        int r = wid * 16 + rr;
        const float* xs = x    + (size_t)s_src[r] * DD;
        const float* rl = relc + (size_t)s_et [r] * DD;
        for (int c = lane; c < CHUNKS; c += 32) {
            int d = c * 4;
            float vx = 0.f, vy = 0.f, vz = 0.f, vw = 0.f;
            if (d < DD) {
                float4 x4 = load_f4(xs + d);
                float4 r4 = load_f4(rl + d);
                float4 q4 = load_bf4f(bufQ + r * LDA + d);
                vx = x4.x * (0.5f * r4.x + 0.5f * q4.x);
                vy = x4.y * (0.5f * r4.y + 0.5f * q4.y);
                vz = x4.z * (0.5f * r4.z + 0.5f * q4.z);
                vw = x4.w * (0.5f * r4.w + 0.5f * q4.w);
            }
            store_bf4(bufB + r * LDA + d, vx, vy, vz, vw);
        }
    }
    __syncthreads();

    // GEMM2: msg = a2 @ W ; scale by norm, scatter-add to acc[dst]
    for (int nt = wid; nt < NTILES; nt += 4) {
        v8f c[MT];
        #pragma unroll
        for (int m = 0; m < MT; ++m) c[m] = v8f{0.f, 0.f, 0.f, 0.f, 0.f, 0.f, 0.f, 0.f};
        v16bf bcur = load_b_frag(wT, nt, lane, 0);
        #pragma unroll
        for (int kt = 0; kt < KTILES; ++kt) {
            v16bf bnext;
            if (kt + 1 < KTILES) bnext = load_b_frag(wT, nt, lane, (kt + 1) * 32);
            #pragma unroll
            for (int m = 0; m < MT; ++m)
                c[m] = wmma_bf16(load_a_frag(bufB + m * 16 * LDA, lane, kt * 32), bcur, c[m]);
            if (kt + 1 < KTILES) bcur = bnext;
        }
        int col = nt * 16 + l16;
        if (col < DD) {
            #pragma unroll
            for (int m = 0; m < MT; ++m) {
                #pragma unroll
                for (int v = 0; v < 8; ++v) {
                    int rr = m * 16 + v + 8 * hf;
                    atomicAdd(&acc[(size_t)s_dst[rr] * DD + col], c[m][v] * s_nrm[rr]);
                }
            }
        }
    }
}

// ---------------- self-loop GEMM fused with (acc+loop)/3 + bias, in place ----------------
__global__ __launch_bounds__(128) void loop_bias_kernel(
    const float* __restrict__ x, const float* __restrict__ lrel,
    const __bf16* __restrict__ wT, const float* __restrict__ bias,
    float* __restrict__ acc)
{
    __shared__ __align__(16) __bf16 bufA[ROWS * LDA];
    const int tid  = threadIdx.x;
    const int wid  = tid >> 5;
    const int lane = tid & 31;
    const int rbase = blockIdx.x * ROWS;           // grid exact: NUM_ENT/64

    for (int rr = 0; rr < 16; ++rr) {
        int r = wid * 16 + rr;
        const float* xr = x + (size_t)(rbase + r) * DD;
        for (int c = lane; c < CHUNKS; c += 32) {
            int d = c * 4;
            float vx = 0.f, vy = 0.f, vz = 0.f, vw = 0.f;
            if (d < DD) {
                float4 x4 = load_f4(xr + d);
                float4 l4 = load_f4(lrel + d);
                vx = x4.x * l4.x; vy = x4.y * l4.y; vz = x4.z * l4.z; vw = x4.w * l4.w;
            }
            store_bf4(bufA + r * LDA + d, vx, vy, vz, vw);
        }
    }
    __syncthreads();

    const int l16 = lane & 15, hf = lane >> 4;
    for (int nt = wid; nt < NTILES; nt += 4) {
        v8f c[MT];
        #pragma unroll
        for (int m = 0; m < MT; ++m) c[m] = v8f{0.f, 0.f, 0.f, 0.f, 0.f, 0.f, 0.f, 0.f};
        v16bf bcur = load_b_frag(wT, nt, lane, 0);
        #pragma unroll
        for (int kt = 0; kt < KTILES; ++kt) {
            v16bf bnext;
            if (kt + 1 < KTILES) bnext = load_b_frag(wT, nt, lane, (kt + 1) * 32);
            #pragma unroll
            for (int m = 0; m < MT; ++m)
                c[m] = wmma_bf16(load_a_frag(bufA + m * 16 * LDA, lane, kt * 32), bcur, c[m]);
            if (kt + 1 < KTILES) bcur = bnext;
        }
        int col = nt * 16 + l16;
        if (col < DD) {
            #pragma unroll
            for (int m = 0; m < MT; ++m) {
                #pragma unroll
                for (int v = 0; v < 8; ++v) {
                    int row = rbase + m * 16 + v + 8 * hf;
                    size_t o = (size_t)row * DD + col;
                    acc[o] = (acc[o] + c[m][v]) * (1.0f / 3.0f) + bias[col];
                }
            }
        }
    }
}

// ---------------- rel update: rout[0:200] = (rel_cur[201x200] @ w_rel) ----------------
__global__ __launch_bounds__(128) void rel_gemm_kernel(
    const float* __restrict__ relc, const __bf16* __restrict__ wT, float* __restrict__ rout)
{
    __shared__ __align__(16) __bf16 bufA[4][16 * LDA];
    const int wid  = threadIdx.x >> 5;
    const int lane = threadIdx.x & 31;
    const int tile = blockIdx.x * 4 + wid;
    if (tile >= NTILES) return;
    const int rbase = tile * 16;
    __bf16* A = bufA[wid];

    for (int rr = 0; rr < 16; ++rr) {
        int row = rbase + rr;
        for (int d = lane; d < KPAD; d += 32) {
            float v = 0.f;
            if (row <= NREL2 && d < DD) v = relc[(size_t)row * DD + d];
            A[rr * LDA + d] = (__bf16)v;
        }
    }
    asm volatile("s_wait_dscnt 0" ::: "memory");

    const int l16 = lane & 15, hf = lane >> 4;
    for (int nt = 0; nt < NTILES; ++nt) {
        v8f c = {0.f, 0.f, 0.f, 0.f, 0.f, 0.f, 0.f, 0.f};
        #pragma unroll
        for (int kt = 0; kt < KTILES; ++kt)
            c = wmma_bf16(load_a_frag(A, lane, kt * 32), load_b_frag(wT, nt, lane, kt * 32), c);
        int col = nt * 16 + l16;
        if (col < DD) {
            #pragma unroll
            for (int v = 0; v < 8; ++v) {
                int row = rbase + v + 8 * hf;
                if (row < NREL2) rout[(size_t)row * DD + col] = c[v];
            }
        }
    }
}

// ---------------- BatchNorm stats: float4 linear pass + per-column LDS partials -------------
__global__ void bn_partial_kernel(const float* __restrict__ a, float* __restrict__ gsum,
                                  float* __restrict__ gsq, int n4) {
    __shared__ float lsum[DD];
    __shared__ float lsq[DD];
    for (int t = threadIdx.x; t < DD; t += blockDim.x) { lsum[t] = 0.f; lsq[t] = 0.f; }
    __syncthreads();
    int stride = gridDim.x * blockDim.x;
    for (int i4 = blockIdx.x * blockDim.x + threadIdx.x; i4 < n4; i4 += stride) {
        float4 v = load_f4(a + (size_t)i4 * 4);
        int col = (i4 * 4) % DD;               // DD % 4 == 0: 4 cols stay in-row
        atomicAdd(&lsum[col + 0], v.x); atomicAdd(&lsq[col + 0], v.x * v.x);
        atomicAdd(&lsum[col + 1], v.y); atomicAdd(&lsq[col + 1], v.y * v.y);
        atomicAdd(&lsum[col + 2], v.z); atomicAdd(&lsq[col + 2], v.z * v.z);
        atomicAdd(&lsum[col + 3], v.w); atomicAdd(&lsq[col + 3], v.w * v.w);
    }
    __syncthreads();
    for (int t = threadIdx.x; t < DD; t += blockDim.x) {
        atomicAdd(&gsum[t], lsum[t]);
        atomicAdd(&gsq[t],  lsq[t]);
    }
}

__global__ void bn_final_kernel(const float* __restrict__ gsum, const float* __restrict__ gsq,
                                float* __restrict__ mean, float* __restrict__ var, int rows) {
    int t = blockIdx.x * blockDim.x + threadIdx.x;
    if (t >= DD) return;
    float m = gsum[t] / (float)rows;
    mean[t] = m;
    var[t]  = gsq[t] / (float)rows - m * m;
}

__global__ void bn_apply_kernel(const float* __restrict__ a, const float* __restrict__ mean,
                                const float* __restrict__ var, const float* __restrict__ gamma,
                                const float* __restrict__ beta, float* __restrict__ out, int n4) {
    int i4 = blockIdx.x * blockDim.x + threadIdx.x;
    if (i4 >= n4) return;
    int col = (i4 * 4) % DD;
    float4 v = load_f4(a + (size_t)i4 * 4);
    float4 o;
    o.x = tanhf(gamma[col + 0] * (v.x - mean[col + 0]) * rsqrtf(var[col + 0] + BN_EPS) + beta[col + 0]);
    o.y = tanhf(gamma[col + 1] * (v.y - mean[col + 1]) * rsqrtf(var[col + 1] + BN_EPS) + beta[col + 1]);
    o.z = tanhf(gamma[col + 2] * (v.z - mean[col + 2]) * rsqrtf(var[col + 2] + BN_EPS) + beta[col + 2]);
    o.w = tanhf(gamma[col + 3] * (v.w - mean[col + 3]) * rsqrtf(var[col + 3] + BN_EPS) + beta[col + 3]);
    *reinterpret_cast<float4*>(out + (size_t)i4 * 4) = o;
}

// ---------------- final gathers (float4) ----------------
__global__ void gather_kernel(const float* __restrict__ x2, const float* __restrict__ relf,
                              const int* __restrict__ sub, const int* __restrict__ rel,
                              float* __restrict__ out_sub, float* __restrict__ out_rel, int n4) {
    int i4 = blockIdx.x * blockDim.x + threadIdx.x;
    if (i4 >= n4) return;
    int i = i4 * 4;
    int b = i / DD, d = i % DD;
    *reinterpret_cast<float4*>(out_sub + i) = load_f4(x2   + (size_t)sub[b] * DD + d);
    *reinterpret_cast<float4*>(out_rel + i) = load_f4(relf + (size_t)rel[b] * DD + d);
}

// ---------------- host-side orchestration ----------------
extern "C" void kernel_launch(void* const* d_in, const int* in_sizes, int n_in,
                              void* d_out, int out_size, void* d_ws, size_t ws_size,
                              hipStream_t stream) {
    const int* ei   = (const int*)d_in[0];     // [2][TOTE]
    const int* etp  = (const int*)d_in[1];     // [TOTE]
    const int* qep  = (const int*)d_in[2];     // [2][TOTE]
    const int* qrp  = (const int*)d_in[3];     // [2][TOTE]
    const int* sub  = (const int*)d_in[4];     // [BQ]
    const int* rel  = (const int*)d_in[5];     // [BQ]
    const float* init_embed = (const float*)d_in[6];
    const float* init_rel   = (const float*)d_in[7];

    const int WLOOP[2] = {8, 17}, WIN[2] = {9, 18}, WOUT[2] = {10, 19};
    const int WREL[2] = {11, 20}, WQ[2] = {12, 21}, LREL[2] = {13, 22};
    const int BIAS[2] = {14, 23}, GAM[2] = {15, 24}, BET[2] = {16, 25};

    char* ws = (char*)d_ws;
    size_t off = 0;
    auto carve = [&](size_t bytes) -> char* {
        off = (off + 255) & ~(size_t)255;
        char* p = ws + off;
        off += bytes;
        return p;
    };
    float*  deg_in   = (float*)carve(NUM_ENT * sizeof(float));
    float*  deg_out  = (float*)carve(NUM_ENT * sizeof(float));
    float*  dinv_in  = (float*)carve(NUM_ENT * sizeof(float));
    float*  dinv_out = (float*)carve(NUM_ENT * sizeof(float));
    float*  rel_cur  = (float*)carve((NREL2 + 1) * DD * sizeof(float));
    float*  rel_next = (float*)carve(NREL2 * DD * sizeof(float));
    float*  acc      = (float*)carve((size_t)NUM_ENT * DD * sizeof(float));
    float*  x1       = (float*)carve((size_t)NUM_ENT * DD * sizeof(float));
    float*  bn_acc   = (float*)carve(512 * sizeof(float));   // gsum[256] | gsq[256]
    float*  bn_mean  = (float*)carve(256 * sizeof(float));
    float*  bn_var   = (float*)carve(256 * sizeof(float));
    __bf16* wt_all   = (__bf16*)carve((size_t)10 * NPAD * KPAD * sizeof(__bf16));
    auto wt = [&](int layer, int j) -> __bf16* {  // 0=w_q 1=w_in 2=w_out 3=w_loop 4=w_rel
        return wt_all + (size_t)(layer * 5 + j) * NPAD * KPAD;
    };
    float* gsum = bn_acc;
    float* gsq  = bn_acc + 256;

    float* out_sub = (float*)d_out;
    float* out_rel = (float*)d_out + (size_t)BQ * DD;
    float* x2      = (float*)d_out + (size_t)2 * BQ * DD;

    // 1) degrees + dinv (layer-invariant)
    hipMemsetAsync(deg_in,  0, NUM_ENT * sizeof(float), stream);
    hipMemsetAsync(deg_out, 0, NUM_ENT * sizeof(float), stream);
    deg_kernel<<<(TOTE + 255) / 256, 256, 0, stream>>>(ei, deg_in, deg_out, EDG);
    dinv_kernel<<<(NUM_ENT + 255) / 256, 256, 0, stream>>>(deg_in, deg_out, dinv_in, dinv_out, NUM_ENT);

    // 2) weight transpose/convert (10 matrices)
    const int wgrid = (NPAD * KPAD + 255) / 256;
    for (int l = 0; l < 2; ++l) {
        wprep_kernel<<<wgrid, 256, 0, stream>>>((const float*)d_in[WQ[l]],    wt(l, 0));
        wprep_kernel<<<wgrid, 256, 0, stream>>>((const float*)d_in[WIN[l]],   wt(l, 1));
        wprep_kernel<<<wgrid, 256, 0, stream>>>((const float*)d_in[WOUT[l]],  wt(l, 2));
        wprep_kernel<<<wgrid, 256, 0, stream>>>((const float*)d_in[WLOOP[l]], wt(l, 3));
        wprep_kernel<<<wgrid, 256, 0, stream>>>((const float*)d_in[WREL[l]],  wt(l, 4));
    }

    // 3) rel_cur for layer 1 = concat(init_rel, l1_loop_rel)
    hipMemcpyAsync(rel_cur, init_rel, (size_t)NREL2 * DD * sizeof(float),
                   hipMemcpyDeviceToDevice, stream);
    hipMemcpyAsync(rel_cur + (size_t)NREL2 * DD, d_in[LREL[0]], DD * sizeof(float),
                   hipMemcpyDeviceToDevice, stream);

    const int edge_blocks = EDG / ROWS;        // 1250
    const int loop_blocks = NUM_ENT / ROWS;    // 625
    const int n4 = NUM_ENT * DD / 4;

    for (int l = 0; l < 2; ++l) {
        const float* xcur  = (l == 0) ? init_embed : x1;
        float*       xnext = (l == 0) ? x1 : x2;

        hipMemsetAsync(acc, 0, (size_t)NUM_ENT * DD * sizeof(float), stream);

        edge_msg_kernel<<<edge_blocks, 128, 0, stream>>>(
            xcur, rel_cur, ei, ei + TOTE, etp, qep, qrp, TOTE,
            dinv_in, wt(l, 0), wt(l, 1), acc);
        edge_msg_kernel<<<edge_blocks, 128, 0, stream>>>(
            xcur, rel_cur, ei + EDG, ei + TOTE + EDG, etp + EDG, qep + EDG, qrp + EDG, TOTE,
            dinv_out, wt(l, 0), wt(l, 2), acc);

        loop_bias_kernel<<<loop_blocks, 128, 0, stream>>>(
            xcur, (const float*)d_in[LREL[l]], wt(l, 3), (const float*)d_in[BIAS[l]], acc);

        hipMemsetAsync(bn_acc, 0, 512 * sizeof(float), stream);
        bn_partial_kernel<<<256, 256, 0, stream>>>(acc, gsum, gsq, n4);
        bn_final_kernel<<<1, 256, 0, stream>>>(gsum, gsq, bn_mean, bn_var, NUM_ENT);
        bn_apply_kernel<<<(n4 + 255) / 256, 256, 0, stream>>>(
            acc, bn_mean, bn_var, (const float*)d_in[GAM[l]], (const float*)d_in[BET[l]],
            xnext, n4);

        rel_gemm_kernel<<<(NTILES + 3) / 4, 128, 0, stream>>>(rel_cur, wt(l, 4), rel_next);

        if (l == 0) {
            hipMemcpyAsync(rel_cur, rel_next, (size_t)NREL2 * DD * sizeof(float),
                           hipMemcpyDeviceToDevice, stream);
            hipMemcpyAsync(rel_cur + (size_t)NREL2 * DD, d_in[LREL[1]], DD * sizeof(float),
                           hipMemcpyDeviceToDevice, stream);
        }
    }

    gather_kernel<<<(BQ * DD / 4 + 255) / 256, 256, 0, stream>>>(
        x2, rel_next, sub, rel, out_sub, out_rel, BQ * DD / 4);
}
